// DSS_69312182223490
// MI455X (gfx1250) — matrix operand order repaired and checked by reading, not compiled, not run on hardware
//
#include <hip/hip_runtime.h>
#include <hip/hip_bf16.h>
#include <stdint.h>

#define BATCH 4
#define SEQL  4096
#define DIM_  1024
#define KN    512          // number of SSM modes
#define QC    256          // chunk length
#define CC    (SEQL/QC)    // 16 chunks
#define LN_EPS 1e-5f

typedef __attribute__((ext_vector_type(16))) __bf16 v16bf;
typedef __attribute__((ext_vector_type(8)))  __bf16 v8bf;
typedef __attribute__((ext_vector_type(8)))  float  v8f;
typedef __attribute__((ext_vector_type(4)))  unsigned int v4u;
typedef __attribute__((ext_vector_type(8)))  int v8i;
typedef __attribute__((ext_vector_type(4)))  int v4i;

#if defined(__has_builtin)
#if __has_builtin(__builtin_amdgcn_tensor_load_to_lds) && __has_builtin(__builtin_amdgcn_s_wait_tensorcnt)
#define HAVE_TDM 1
#endif
#endif
#ifndef HAVE_TDM
#define HAVE_TDM 0
#endif

union Frag { v16bf v; v8bf h[2]; };

#define BM 128
#define BN 128
#define BK 32
#define SA 48   // LDS row stride for A tile (bf16), keeps 16B alignment

// ---------------------------------------------------------------- LayerNorm
__global__ __launch_bounds__(256) void ln_kernel(const float* __restrict__ x,
                                                 const float* __restrict__ g,
                                                 const float* __restrict__ be,
                                                 float* __restrict__ u) {
  __shared__ float s1[256], s2[256];
  const int row = blockIdx.x;
  const int t = threadIdx.x;
  const float* xr = x + (size_t)row * DIM_;
  float v[4]; float sum = 0.f, sq = 0.f;
#pragma unroll
  for (int k = 0; k < 4; ++k) { v[k] = xr[t + k*256]; sum += v[k]; sq += v[k]*v[k]; }
  s1[t] = sum; s2[t] = sq; __syncthreads();
  for (int o = 128; o > 0; o >>= 1) {
    if (t < o) { s1[t] += s1[t+o]; s2[t] += s2[t+o]; }
    __syncthreads();
  }
  const float mu  = s1[0] * (1.f/DIM_);
  const float var = s2[0] * (1.f/DIM_) - mu*mu;
  const float rs  = rsqrtf(var + LN_EPS);
  float* ur = u + (size_t)row * DIM_;
#pragma unroll
  for (int k = 0; k < 4; ++k) { int d = t + k*256; ur[d] = (v[k]-mu)*rs*g[d] + be[d]; }
}

// ------------------------------------------------- Lambda -> (a, theta)
__global__ void prep_lambda(const float* __restrict__ Lr, const float* __restrict__ Li,
                            float* __restrict__ a, float* __restrict__ th) {
  int n = blockIdx.x*blockDim.x + threadIdx.x;
  if (n >= KN) return;
  a[n]  = -__expf(Lr[n]);
  th[n] =  __expf(Li[n]);
}

// E[n][m] = exp(Lambda_n * m), m = 0..QC  (real/imag)
__global__ void build_E(const float* __restrict__ a, const float* __restrict__ th,
                        float* __restrict__ Er, float* __restrict__ Ei) {
  int idx = blockIdx.x*256 + threadIdx.x;
  if (idx >= KN*(QC+1)) return;
  int n = idx / (QC+1), m = idx % (QC+1);
  float e  = __expf(a[n] * (float)m);
  float ph = th[n] * (float)m;
  Er[idx] = e * __cosf(ph);
  Ei[idx] = e * __sinf(ph);
}

// A2 [2KN x QC] (state gemm A), B2 [QC x 2KN] (cross gemm A), bf16
__global__ void build_mats(const float* __restrict__ Er, const float* __restrict__ Ei,
                           __bf16* __restrict__ A2, __bf16* __restrict__ B2) {
  int idx = blockIdx.x*256 + threadIdx.x;
  if (idx >= KN*QC) return;
  int n = idx / QC, i = idx % QC;
  // state:  exp(Lambda*(Q-1-i))
  A2[(size_t)n*QC + i]        = (__bf16)Er[n*(QC+1) + (QC-1-i)];
  A2[(size_t)(KN+n)*QC + i]   = (__bf16)Ei[n*(QC+1) + (QC-1-i)];
  // cross:  exp(Lambda*(j+1)),  imag negated (take real part of Ct*P*exp)
  B2[(size_t)i*(2*KN) + n]      = (__bf16)Er[n*(QC+1) + i + 1];
  B2[(size_t)i*(2*KN) + KN + n] = (__bf16)(-Ei[n*(QC+1) + i + 1]);
}

// Ct[d,n] = (C_r + iC_i) * (exp(Lambda)-1)/Lambda
__global__ void ct_kernel(const float* __restrict__ Cr, const float* __restrict__ Ci,
                          const float* __restrict__ a, const float* __restrict__ th,
                          const float* __restrict__ Er, const float* __restrict__ Ei,
                          float* __restrict__ Ctr, float* __restrict__ Cti) {
  int idx = blockIdx.x*256 + threadIdx.x;
  if (idx >= DIM_*KN) return;
  int n = idx % KN;
  float A = a[n], T = th[n];
  float nr = Er[n*(QC+1)+1] - 1.f, ni = Ei[n*(QC+1)+1];
  float inv = 1.f / (A*A + T*T);
  float gr = (nr*A + ni*T) * inv;
  float gi = (ni*A - nr*T) * inv;
  float cr = Cr[idx], ci = Ci[idx];
  Ctr[idx] = cr*gr - ci*gi;
  Cti[idx] = cr*gi + ci*gr;
}

// K[d][m] = Re sum_n Ct[d,n] exp(Lambda_n m),  m < QC
__global__ __launch_bounds__(256) void kshort(const float* __restrict__ Ctr,
                                              const float* __restrict__ Cti,
                                              const float* __restrict__ Er,
                                              const float* __restrict__ Ei,
                                              float* __restrict__ Kd) {
  int idx = blockIdx.x*256 + threadIdx.x;
  if (idx >= DIM_*QC) return;
  int d = idx / QC, m = idx % QC;
  float acc = 0.f;
  for (int n = 0; n < KN; ++n)
    acc += Ctr[(size_t)d*KN+n]*Er[n*(QC+1)+m] - Cti[(size_t)d*KN+n]*Ei[n*(QC+1)+m];
  Kd[idx] = acc;
}

// ------------------------------------------------- TDM tile load (A operand)
#if HAVE_TDM
__device__ __forceinline__ void tdm_load_a_tile(const __bf16* gsrc, unsigned ldsByteOff,
                                                int ldK, int rowsTotal) {
  unsigned long long ga = (unsigned long long)(uintptr_t)gsrc;
  v4u g0;
  g0.x = 1u;                                                    // count=1 (valid D#)
  g0.y = ldsByteOff;                                            // lds_addr
  g0.z = (unsigned)ga;                                          // global_addr[31:0]
  g0.w = (unsigned)((ga >> 32) & 0x1FFFFFFull) | 0x80000000u;   // addr[56:32] | type=2
  v8i g1;
  // data_size=2B (code 1), pad_enable, pad_interval=16 DWORDs (one 64B row),
  // pad_amount=8 DWORDs (32B) -> LDS row stride 96B = 48 bf16 (= SA)
  g1[0] = (int)((1u << 16) | (1u << 20) | (3u << 22) | (7u << 25));
  g1[1] = (int)(((unsigned)ldK & 0xFFFFu) << 16);                        // tensor_dim0 lo
  g1[2] = (int)((((unsigned)ldK >> 16) & 0xFFFFu) |
                (((unsigned)rowsTotal & 0xFFFFu) << 16));                // dim0 hi | dim1 lo
  g1[3] = (int)((((unsigned)rowsTotal >> 16) & 0xFFFFu) | (32u << 16));  // dim1 hi | tile_dim0=32
  g1[4] = (int)128;                                                      // tile_dim1=128, tile_dim2=0
  g1[5] = (int)ldK;                                                      // tensor_dim0_stride lo
  g1[6] = 0;
  g1[7] = 0;
  v4i gz = {0, 0, 0, 0};
#if __clang_major__ >= 23
  v8i gz8 = {0, 0, 0, 0, 0, 0, 0, 0};
  __builtin_amdgcn_tensor_load_to_lds(g0, g1, gz, gz, gz8, 0);
#else
  __builtin_amdgcn_tensor_load_to_lds(g0, g1, gz, gz, 0);
#endif
}
#endif

// --------------------------------------- vectorized B-tile staging (transpose)
// fp32 source: each thread loads one float4 (16B) per step, 4 steps
__device__ __forceinline__ void stage_b_tile(__bf16* Bl, const float* bptr,
                                             long rowStep, int t) {
  const int nq = (t & 31) * 4;   // n within tile: 0..124 (4 consecutive)
  float4 v[4];
#pragma unroll
  for (int r = 0; r < 4; ++r)
    v[r] = *(const float4*)(bptr + (size_t)r * 8 * rowStep);
  const int kb = t >> 5;         // k: kb + 8r
#pragma unroll
  for (int r = 0; r < 4; ++r) {
    int k = kb + r*8;
    Bl[(nq+0)*BK + k] = (__bf16)v[r].x;
    Bl[(nq+1)*BK + k] = (__bf16)v[r].y;
    Bl[(nq+2)*BK + k] = (__bf16)v[r].z;
    Bl[(nq+3)*BK + k] = (__bf16)v[r].w;
  }
}
// bf16 source: each thread loads one v8bf (16B) per step, 2 steps
__device__ __forceinline__ void stage_b_tile(__bf16* Bl, const __bf16* bptr,
                                             long rowStep, int t) {
  const int nq = (t & 15) * 8;   // n within tile: 0..120 (8 consecutive)
  v8bf v[2];
#pragma unroll
  for (int r = 0; r < 2; ++r)
    v[r] = *(const v8bf*)(bptr + (size_t)r * 16 * rowStep);
  const int kb = t >> 4;         // k: kb + 16r
#pragma unroll
  for (int r = 0; r < 2; ++r) {
    int k = kb + r*16;
#pragma unroll
    for (int j = 0; j < 8; ++j)
      Bl[(nq+j)*BK + k] = v[r][j];
  }
}
__device__ __forceinline__ long stage_b_off(const float*, int N, int n0, int t) {
  return (long)(t >> 5) * N + n0 + (t & 31) * 4;
}
__device__ __forceinline__ long stage_b_off(const __bf16*, int N, int n0, int t) {
  return (long)(t >> 4) * N + n0 + (t & 15) * 8;
}

// ------------------------------------------------- batched WMMA GEMM
// C[z] = A (bf16, shared, [M x K] row-major) @ B[z] ([K x N], BT -> bf16)
template <typename BT, typename CT>
__global__ __launch_bounds__(256) void wmma_gemm(const __bf16* __restrict__ A,
                                                 const BT* __restrict__ B,
                                                 CT* __restrict__ C,
                                                 int M, int K, int N,
                                                 long strideB, long strideC) {
  __shared__ __align__(16) __bf16 Al[BM * SA];
  __shared__ __align__(16) __bf16 Bl[BN * BK];   // [n][k]
  CT* Cb = C + (size_t)blockIdx.z * strideC;
  const int m0 = blockIdx.y * BM, n0 = blockIdx.x * BN;
  const int t = threadIdx.x;
  const int lane = t & 31, w = t >> 5;
  const int wm = w >> 2, wn = w & 3;           // 2 x 4 wave grid
  const int lo = lane & 15, hi = lane >> 4;

  // per-thread running global pointer into B (advances by BK rows per K-step)
  const BT* bptr = B + (size_t)blockIdx.z * strideB + stage_b_off((const BT*)0, N, n0, t);

  v8f acc[4][2];
#pragma unroll
  for (int mt = 0; mt < 4; ++mt)
#pragma unroll
    for (int nt = 0; nt < 2; ++nt) acc[mt][nt] = (v8f)0.f;

  for (int k0 = 0; k0 < K; k0 += BK) {
#if HAVE_TDM
    if (w == 0)                                  // one TDM op per workgroup
      tdm_load_a_tile(A + (size_t)m0 * K + k0,
                      (unsigned)(uintptr_t)(void*)&Al[0], K, M);
#else
#pragma unroll
    for (int r = 0; r < 16; ++r) {               // manual A staging fallback
      int i = r*256 + t; int m = i >> 5, k = i & 31;
      Al[m*SA + k] = A[(size_t)(m0+m)*K + k0 + k];
    }
#endif
    stage_b_tile(Bl, bptr, (long)N, t);          // vectorized convert+transpose
    if (k0 + BK < K)                             // prefetch next B tile
      __builtin_prefetch((const void*)(bptr + (size_t)BK * N), 0, 1);
    bptr += (size_t)BK * N;
#if HAVE_TDM
    if (w == 0) __builtin_amdgcn_s_wait_tensorcnt(0);
#endif
    __syncthreads();

    Frag af[4], bfr[2];
#pragma unroll
    for (int mt = 0; mt < 4; ++mt) {
      const __bf16* row = &Al[(wm*64 + mt*16 + lo) * SA];
      af[mt].h[0] = *(const v8bf*)(row + hi*8);
      af[mt].h[1] = *(const v8bf*)(row + 16 + hi*8);
    }
#pragma unroll
    for (int nt = 0; nt < 2; ++nt) {
      const __bf16* row = &Bl[(wn*32 + nt*16 + lo) * BK];
      bfr[nt].h[0] = *(const v8bf*)(row + hi*8);
      bfr[nt].h[1] = *(const v8bf*)(row + 16 + hi*8);
    }
#pragma unroll
    for (int mt = 0; mt < 4; ++mt)
#pragma unroll
      for (int nt = 0; nt < 2; ++nt)
        acc[mt][nt] = __builtin_amdgcn_wmma_f32_16x16x32_bf16(
            false, af[mt].v, false, bfr[nt].v, (short)0, acc[mt][nt], false, false);
    __syncthreads();
  }

#pragma unroll
  for (int mt = 0; mt < 4; ++mt)
#pragma unroll
    for (int nt = 0; nt < 2; ++nt) {
      CT* crow = Cb + (size_t)(m0 + wm*64 + mt*16 + 8*hi) * N
                    + (n0 + wn*32 + nt*16 + lo);
#pragma unroll
      for (int r = 0; r < 8; ++r)
        crow[(size_t)r * N] = (CT)acc[mt][nt][r];
    }
}

// ---------------------------------------------- inter-chunk recurrence (in place, bf16)
// Tbuf layout: [b][c][2KN][DIM]; rows n = real, KN+n = imag.
// Replaces T with R = Ct * P (P = state entering chunk); fp32 register math.
__global__ __launch_bounds__(256) void chunk_scan(__bf16* __restrict__ T,
                                                  const float* __restrict__ Ctr,
                                                  const float* __restrict__ Cti,
                                                  const float* __restrict__ Er,
                                                  const float* __restrict__ Ei) {
  size_t tid = (size_t)blockIdx.x*256 + threadIdx.x;      // B * KN * DIM threads
  int d = (int)(tid % DIM_);
  int n = (int)((tid / DIM_) % KN);
  int b = (int)(tid / ((size_t)DIM_ * KN));
  const float wr = Er[n*(QC+1) + QC], wi = Ei[n*(QC+1) + QC];  // exp(Lambda*Q)
  const float cr = Ctr[(size_t)d*KN + n], ci = Cti[(size_t)d*KN + n];
  float pr = 0.f, pi = 0.f;
  for (int c = 0; c < CC; ++c) {
    size_t base = ((size_t)(b*CC + c)*(2*KN) + n)*DIM_ + d;
    float tr = (float)T[base];
    float ti = (float)T[base + (size_t)KN*DIM_];
    T[base]                   = (__bf16)(cr*pr - ci*pi);    // R real
    T[base + (size_t)KN*DIM_] = (__bf16)(cr*pi + ci*pr);    // R imag
    float npr = wr*pr - wi*pi + tr;
    float npi = wr*pi + wi*pr + ti;
    pr = npr; pi = npi;
  }
}

// ------------------------------- intra-chunk causal conv + residual (y += ...)
#define DT 16
__global__ __launch_bounds__(256) void intra_residual(const float* __restrict__ u,
                                                      const float* __restrict__ Kd,
                                                      const float* __restrict__ pD,
                                                      float* __restrict__ y) {
  __shared__ float us[QC*(DT+1)];   // [i][dd]
  __shared__ float Ks[QC*(DT+1)];   // [m][dd]
  const int bc = blockIdx.x;        // b*CC + c
  const int d0 = blockIdx.y * DT;
  const int t = threadIdx.x;
#pragma unroll
  for (int r = 0; r < (QC*DT)/256; ++r) {      // stage u chunk tile
    int i = r*256 + t; int ii = i / DT, dd = i % DT;
    us[ii*(DT+1) + dd] = u[((size_t)bc*QC + ii)*DIM_ + d0 + dd];
  }
#pragma unroll
  for (int r = 0; r < (QC*DT)/256; ++r) {      // stage K tile (coalesced over m)
    int i = r*256 + t; int m = i % QC, dd = i / QC;
    Ks[m*(DT+1) + dd] = Kd[(size_t)(d0+dd)*QC + m];
  }
  __syncthreads();
  const int dd = t % DT, js = t / DT;
  const float pdv = pD[d0 + dd];
  for (int jj = js; jj < QC; jj += 256/DT) {
    float acc = 0.f;
    for (int i = 0; i <= jj; ++i)
      acc += Ks[(jj-i)*(DT+1) + dd] * us[i*(DT+1) + dd];
    size_t g = ((size_t)bc*QC + jj)*DIM_ + d0 + dd;
    y[g] += acc + us[jj*(DT+1) + dd] * pdv;    // + residual u*param_D
  }
}

// ----------------------------------------------------------------- launcher
extern "C" void kernel_launch(void* const* d_in, const int* in_sizes, int n_in,
                              void* d_out, int out_size, void* d_ws, size_t ws_size,
                              hipStream_t stream) {
  const float* x   = (const float*)d_in[0];
  const float* Lr  = (const float*)d_in[1];
  const float* Li  = (const float*)d_in[2];
  const float* Cr  = (const float*)d_in[3];
  const float* Ci  = (const float*)d_in[4];
  const float* pD  = (const float*)d_in[5];
  const float* lg  = (const float*)d_in[6];
  const float* lb  = (const float*)d_in[7];
  float* out = (float*)d_out;

  char* ws = (char*)d_ws;
  const size_t szU  = (size_t)BATCH*SEQL*DIM_*4;             // 64 MB
  const size_t szT  = (size_t)BATCH*CC*2*KN*DIM_*2;          // 128 MB (bf16)
  const size_t szCt = (size_t)DIM_*KN*4;                     // 2 MB
  const size_t szE  = (size_t)KN*(QC+1)*4;
  const size_t szK  = (size_t)DIM_*QC*4;
  const size_t szM  = (size_t)2*KN*QC*2;                     // bf16 mats

  float*  u    = (float*)(ws);
  __bf16* Tbuf = (__bf16*)(ws + szU);
  float*  Ctr  = (float*)(ws + szU + szT);
  float*  Cti  = (float*)(ws + szU + szT + szCt);
  float*  Er   = (float*)(ws + szU + szT + 2*szCt);
  float*  Ei   = (float*)(ws + szU + szT + 2*szCt + szE);
  float*  Kd   = (float*)(ws + szU + szT + 2*szCt + 2*szE);
  __bf16* A2   = (__bf16*)(ws + szU + szT + 2*szCt + 2*szE + szK);
  __bf16* B2   = (__bf16*)(ws + szU + szT + 2*szCt + 2*szE + szK + szM);
  float*  av   = (float*)(ws + szU + szT + 2*szCt + 2*szE + szK + 2*szM);
  float*  th   = av + KN;

  prep_lambda<<<(KN+255)/256, 256, 0, stream>>>(Lr, Li, av, th);
  build_E<<<(KN*(QC+1)+255)/256, 256, 0, stream>>>(av, th, Er, Ei);
  build_mats<<<(KN*QC)/256, 256, 0, stream>>>(Er, Ei, A2, B2);
  ct_kernel<<<(DIM_*KN)/256, 256, 0, stream>>>(Cr, Ci, av, th, Er, Ei, Ctr, Cti);
  kshort<<<(DIM_*QC)/256, 256, 0, stream>>>(Ctr, Cti, Er, Ei, Kd);
  ln_kernel<<<BATCH*SEQL, 256, 0, stream>>>(x, lg, lb, u);

  // state GEMM: T[bc] (bf16) = A2[2KN x QC] @ u_chunk[QC x DIM]
  wmma_gemm<float, __bf16><<<dim3(DIM_/BN, (2*KN)/BM, BATCH*CC), 256, 0, stream>>>(
      A2, u, Tbuf, 2*KN, QC, DIM_, (long)QC*DIM_, (long)2*KN*DIM_);

  chunk_scan<<<(unsigned)((size_t)BATCH*KN*DIM_/256), 256, 0, stream>>>(Tbuf, Ctr, Cti, Er, Ei);

  // cross GEMM: y_cross[bc] (f32, d_out) = B2[QC x 2KN] @ R[2KN x DIM] (bf16)
  wmma_gemm<__bf16, float><<<dim3(DIM_/BN, QC/BM, BATCH*CC), 256, 0, stream>>>(
      B2, Tbuf, out, QC, 2*KN, DIM_, (long)2*KN*DIM_, (long)QC*DIM_);

  // intra-chunk conv + residual (accumulates into d_out)
  intra_residual<<<dim3(BATCH*CC, DIM_/DT), 256, 0, stream>>>(u, Kd, pD, out);
}
